// VQTransAE_32323923869722
// MI455X (gfx1250) — matrix operand order repaired and checked by reading, not compiled, not run on hardware
//
#include <hip/hip_runtime.h>
#include <hip/hip_bf16.h>
#include <math.h>

typedef __attribute__((ext_vector_type(16))) _Float16 v16h;
typedef __attribute__((ext_vector_type(8)))  float    v8f;
typedef int v4i_async __attribute__((vector_size(16)));   // pointee type for async-LDS builtin

#define ACT_NONE 0
#define ACT_RELU 1
#define ACT_GELU 2

// CDNA5 async global->LDS copy availability (device pass only).
#if defined(__gfx1250__) && __has_builtin(__builtin_amdgcn_global_load_async_to_lds_b128)
#define HAVE_ASYNC_LDS 1
#else
#define HAVE_ASYNC_LDS 0
#endif

__device__ __forceinline__ void wait_async_lds() {
#if defined(__gfx1250__)
#if __has_builtin(__builtin_amdgcn_s_wait_asynccnt)
  __builtin_amdgcn_s_wait_asynccnt(0);
#else
  asm volatile("s_wait_asynccnt 0x0" ::: "memory");
#endif
#endif
}

__device__ __forceinline__ void async_copy16(const float* gp, float* lp) {
#if HAVE_ASYNC_LDS
  __builtin_amdgcn_global_load_async_to_lds_b128(
      (__attribute__((address_space(1))) v4i_async*)gp,
      (__attribute__((address_space(3))) v4i_async*)lp, 0, 0);
#else
  *(float4*)lp = *(const float4*)gp;
#endif
}

__device__ __forceinline__ float sigmoidf_(float x) { return 1.f / (1.f + __expf(-x)); }
__device__ __forceinline__ float geluf_(float x)    { return 0.5f * x * (1.f + erff(x * 0.70710678118654752f)); }

// Convert 16 contiguous fp32 (at p[0..15]) into a v16h A-fragment half.
__device__ __forceinline__ void cvt_a8(const float* p, v16h& f) {
  const float4 a0 = *(const float4*)(p);
  const float4 a1 = *(const float4*)(p + 4);
  const float4 a2 = *(const float4*)(p + 8);
  const float4 a3 = *(const float4*)(p + 12);
  f[0]=(_Float16)a0.x; f[1]=(_Float16)a0.y; f[2]=(_Float16)a0.z; f[3]=(_Float16)a0.w;
  f[4]=(_Float16)a1.x; f[5]=(_Float16)a1.y; f[6]=(_Float16)a1.z; f[7]=(_Float16)a1.w;
  f[8]=(_Float16)a2.x; f[9]=(_Float16)a2.y; f[10]=(_Float16)a2.z; f[11]=(_Float16)a2.w;
  f[12]=(_Float16)a3.x; f[13]=(_Float16)a3.y; f[14]=(_Float16)a3.z; f[15]=(_Float16)a3.w;
}
// Convert 16 contiguous fp32 into a v16h B-fragment half.
__device__ __forceinline__ void cvt_b16(const float* p, v16h& f) {
  const float4 b0 = *(const float4*)(p);
  const float4 b1 = *(const float4*)(p + 4);
  const float4 b2 = *(const float4*)(p + 8);
  const float4 b3 = *(const float4*)(p + 12);
  f[0]=(_Float16)b0.x; f[1]=(_Float16)b0.y; f[2]=(_Float16)b0.z; f[3]=(_Float16)b0.w;
  f[4]=(_Float16)b1.x; f[5]=(_Float16)b1.y; f[6]=(_Float16)b1.z; f[7]=(_Float16)b1.w;
  f[8]=(_Float16)b2.x; f[9]=(_Float16)b2.y; f[10]=(_Float16)b2.z; f[11]=(_Float16)b2.w;
  f[12]=(_Float16)b3.x; f[13]=(_Float16)b3.y; f[14]=(_Float16)b3.z; f[15]=(_Float16)b3.w;
}

// ---------------------------------------------------------------------------
// Per-wave WMMA GEMM (fallback for small/odd shapes, and the !TRANSB case).
// ---------------------------------------------------------------------------
template <bool TRANSB, int ACT>
__global__ __launch_bounds__(32)
void gemm_wmma(const float* __restrict__ A, int lda, long sAb, long sAh,
               const float* __restrict__ Bm, int ldb, long sBb, long sBh,
               float* __restrict__ C, int ldc, long sCb, long sCh,
               const float* __restrict__ bias,
               const float* __restrict__ resid, int ldr, long sRb, long sRh,
               float scale, int K, int Hb)
{
  const int lane = threadIdx.x;
  const int l15  = lane & 15;
  const int hlf  = lane >> 4;
  const int n0   = blockIdx.x * 32;
  const int m0   = blockIdx.y * 32;
  const long zb  = blockIdx.z / Hb;
  const long zh  = blockIdx.z % Hb;
  A  += zb * sAb + zh * sAh;
  Bm += zb * sBb + zh * sBh;
  C  += zb * sCb + zh * sCh;
  if (resid) resid += zb * sRb + zh * sRh;

  v8f acc[2][2] = {};

  for (int k0 = 0; k0 < K; k0 += 32) {
    v16h af[2], bf[2];
#pragma unroll
    for (int mt = 0; mt < 2; ++mt) {
      const float* ap = A + (long)(m0 + mt * 16 + l15) * lda + k0 + (hlf ? 16 : 0);
      if (k0 + 32 < K) __builtin_prefetch(ap + 32, 0, 1);
      cvt_a8(ap, af[mt]);
    }
#pragma unroll
    for (int nt = 0; nt < 2; ++nt) {
      const int n  = n0 + nt * 16 + l15;
      const int kb = k0 + (hlf ? 16 : 0);
      if (TRANSB) {
        cvt_b16(Bm + (long)n * ldb + kb, bf[nt]);
      } else {
        const float* bp = Bm + (long)kb * ldb + n;
#pragma unroll
        for (int i = 0; i < 16; ++i) bf[nt][i] = (_Float16)bp[(long)i * ldb];
      }
    }
#pragma unroll
    for (int mt = 0; mt < 2; ++mt)
#pragma unroll
      for (int nt = 0; nt < 2; ++nt)
        acc[mt][nt] = __builtin_amdgcn_wmma_f32_16x16x32_f16(
            false, af[mt], false, bf[nt], (short)0, acc[mt][nt], false, false);
  }

#pragma unroll
  for (int mt = 0; mt < 2; ++mt) {
#pragma unroll
    for (int nt = 0; nt < 2; ++nt) {
      const int n  = n0 + nt * 16 + l15;
      const float bv = bias ? bias[n] : 0.f;
#pragma unroll
      for (int r = 0; r < 8; ++r) {
        const int m = m0 + mt * 16 + hlf * 8 + r;
        float v = acc[mt][nt][r] * scale + bv;
        if (resid) v += resid[(long)m * ldr + n];
        if (ACT == ACT_RELU) v = fmaxf(v, 0.f);
        if (ACT == ACT_GELU) v = geluf_(v);
        C[(long)m * ldc + n] = v;
      }
    }
  }
}

// ---------------------------------------------------------------------------
// LDS-staged WMMA GEMM (TRANSB only): 256 threads / 8 waves, 128x64 macro-tile,
// double-buffered fp32 panels staged with async global->LDS copies (ASYNCcnt).
// Requires M%128==0, N%64==0, K%32==0.
// ---------------------------------------------------------------------------
template <int ACT>
__global__ __launch_bounds__(256)
void gemm_wmma_lds(const float* __restrict__ A, int lda, long sAb, long sAh,
                   const float* __restrict__ Bm, int ldb, long sBb, long sBh,
                   float* __restrict__ C, int ldc, long sCb, long sCh,
                   const float* __restrict__ bias,
                   const float* __restrict__ resid, int ldr, long sRb, long sRh,
                   float scale, int K, int Hb)
{
  __shared__ __align__(16) float As[2][128 * 32];
  __shared__ __align__(16) float Bs[2][64 * 32];

  const int tid  = threadIdx.x;
  const int lane = tid & 31;
  const int wave = tid >> 5;         // 8 waves: 4 (M) x 2 (N)
  const int l15  = lane & 15;
  const int hlf  = lane >> 4;
  const int mw   = wave >> 1;
  const int nw   = wave & 1;
  const int n0   = blockIdx.x * 64;
  const int m0   = blockIdx.y * 128;
  const long zb  = blockIdx.z / Hb;
  const long zh  = blockIdx.z % Hb;
  A  += zb * sAb + zh * sAh;
  Bm += zb * sBb + zh * sBh;
  C  += zb * sCb + zh * sCh;
  if (resid) resid += zb * sRb + zh * sRh;

  const int nk = K >> 5;

  auto stage = [&](int kc, int buf) {
    const int k0 = kc * 32;
#pragma unroll
    for (int r = 0; r < 4; ++r) {                 // A panel 128x32 fp32
      const int idx = tid + r * 256;
      const int row = idx >> 3, c4 = (idx & 7) * 4;
      async_copy16(A + (long)(m0 + row) * lda + k0 + c4, &As[buf][row * 32 + c4]);
    }
#pragma unroll
    for (int r = 0; r < 2; ++r) {                 // B panel 64x32 fp32 (rows=N)
      const int idx = tid + r * 256;
      const int row = idx >> 3, c4 = (idx & 7) * 4;
      async_copy16(Bm + (long)(n0 + row) * ldb + k0 + c4, &Bs[buf][row * 32 + c4]);
    }
  };

  v8f acc[2][2] = {};

  stage(0, 0);
#if HAVE_ASYNC_LDS
  wait_async_lds();
#endif
  __syncthreads();

  for (int kc = 0; kc < nk; ++kc) {
    const int cur = kc & 1;
    if (kc + 1 < nk) stage(kc + 1, 1 - cur);   // overlap next panel with math

    const float* as = As[cur];
    const float* bs = Bs[cur];
    v16h af[2], bf[2];
#pragma unroll
    for (int mt = 0; mt < 2; ++mt)
      cvt_a8(as + (mw * 32 + mt * 16 + l15) * 32 + (hlf ? 16 : 0), af[mt]);
#pragma unroll
    for (int nt = 0; nt < 2; ++nt)
      cvt_b16(bs + (nw * 32 + nt * 16 + l15) * 32 + (hlf ? 16 : 0), bf[nt]);
#pragma unroll
    for (int mt = 0; mt < 2; ++mt)
#pragma unroll
      for (int nt = 0; nt < 2; ++nt)
        acc[mt][nt] = __builtin_amdgcn_wmma_f32_16x16x32_f16(
            false, af[mt], false, bf[nt], (short)0, acc[mt][nt], false, false);

#if HAVE_ASYNC_LDS
    wait_async_lds();
#endif
    __syncthreads();
  }

#pragma unroll
  for (int mt = 0; mt < 2; ++mt) {
#pragma unroll
    for (int nt = 0; nt < 2; ++nt) {
      const int n  = n0 + nw * 32 + nt * 16 + l15;
      const float bv = bias ? bias[n] : 0.f;
#pragma unroll
      for (int r = 0; r < 8; ++r) {
        const int m = m0 + mw * 32 + mt * 16 + hlf * 8 + r;
        float v = acc[mt][nt][r] * scale + bv;
        if (resid) v += resid[(long)m * ldr + n];
        if (ACT == ACT_RELU) v = fmaxf(v, 0.f);
        if (ACT == ACT_GELU) v = geluf_(v);
        C[(long)m * ldc + n] = v;
      }
    }
  }
}

// ---------------------------------------------------------------------------
// Persistent LSTM scan: one 1024-thread workgroup (32 waves), h/c/gates in LDS.
// ---------------------------------------------------------------------------
__global__ __launch_bounds__(1024)
void lstm_scan(const float* __restrict__ xs, const float* __restrict__ Whh,
               float* __restrict__ out, int reverse)
{
  __shared__ float    cS[2048];
  __shared__ _Float16 hF[2048];
  __shared__ float    gS[16 * 512];
  const int tid = threadIdx.x;
  for (int i = tid; i < 2048; i += 1024) { cS[i] = 0.f; hF[i] = (_Float16)0.f; }
  __syncthreads();

  const int lane = tid & 31;
  const int wave = tid >> 5;
  const int l15  = lane & 15;
  const int hlf  = lane >> 4;
  const int n0   = wave * 16;

  for (int s = 0; s < 512; ++s) {
    const int t = reverse ? (511 - s) : s;
    v8f acc = {};
#pragma unroll
    for (int kc = 0; kc < 4; ++kc) {
      const int k0 = kc * 32;
      v16h af, bf;
      const _Float16* ap = hF + l15 * 128 + k0 + (hlf ? 16 : 0);
#pragma unroll
      for (int i = 0; i < 8; ++i) { af[i] = ap[i]; af[i + 8] = ap[i + 8]; }
      cvt_b16(Whh + (long)(n0 + l15) * 128 + k0 + (hlf ? 16 : 0), bf);
      acc = __builtin_amdgcn_wmma_f32_16x16x32_f16(false, af, false, bf,
                                                   (short)0, acc, false, false);
    }
    const int n = n0 + l15;
#pragma unroll
    for (int r = 0; r < 8; ++r) {
      const int m = hlf * 8 + r;
      gS[m * 512 + n] = acc[r] + xs[((long)m * 512 + t) * 512 + n];
    }
    __syncthreads();
    for (int cell = tid; cell < 2048; cell += 1024) {
      const int m = cell >> 7, j = cell & 127;
      const float ig = gS[m * 512 + j];
      const float fg = gS[m * 512 + 128 + j];
      const float gg = gS[m * 512 + 256 + j];
      const float og = gS[m * 512 + 384 + j];
      const float c  = sigmoidf_(fg) * cS[cell] + sigmoidf_(ig) * tanhf(gg);
      const float h  = sigmoidf_(og) * tanhf(c);
      cS[cell] = c;
      hF[cell] = (_Float16)h;
      out[((long)m * 512 + t) * 256 + j] = h;
    }
    __syncthreads();
  }
}

// ---------------------------------------------------------------------------
// VQ argmin + quantize (+ per-row distance for deterministic loss reduce).
// ---------------------------------------------------------------------------
__global__ __launch_bounds__(256)
void vq_kernel(const float* __restrict__ ze, const float* __restrict__ E,
               float* __restrict__ zq, int* __restrict__ idxi,
               float* __restrict__ idxf, float* __restrict__ bdist)
{
  __shared__ float zrow[64];
  __shared__ float bd[256];
  __shared__ int   bi[256];
  const int row = blockIdx.x;
  const int tid = threadIdx.x;
  if (tid < 64) zrow[tid] = ze[(long)row * 64 + tid];
  __syncthreads();
  float best = 3.4e38f; int bidx = 0;
  for (int k = tid; k < 1024; k += 256) {
    const float* e = E + (long)k * 64;
    float d = 0.f;
#pragma unroll 8
    for (int j = 0; j < 64; ++j) { float t = zrow[j] - e[j]; d += t * t; }
    if (d < best) { best = d; bidx = k; }
  }
  bd[tid] = best; bi[tid] = bidx; __syncthreads();
  for (int st = 128; st > 0; st >>= 1) {
    if (tid < st) {
      if (bd[tid + st] < bd[tid] ||
          (bd[tid + st] == bd[tid] && bi[tid + st] < bi[tid])) {
        bd[tid] = bd[tid + st]; bi[tid] = bi[tid + st];
      }
    }
    __syncthreads();
  }
  const int k = bi[0];
  if (tid < 64) zq[(long)row * 64 + tid] = E[(long)k * 64 + tid];
  if (tid == 0) { idxi[row] = k; idxf[row] = (float)k; bdist[row] = bd[0]; }
}

__global__ __launch_bounds__(1024)
void vq_loss_reduce(const float* __restrict__ bdist, float* __restrict__ lossp)
{
  __shared__ float red[1024];
  const int tid = threadIdx.x;
  float s = 0.f;
  for (int i = tid; i < 8192; i += 1024) s += bdist[i];
  red[tid] = s; __syncthreads();
  for (int st = 512; st > 0; st >>= 1) {
    if (tid < st) red[tid] += red[tid + st];
    __syncthreads();
  }
  if (tid == 0) *lossp = 1.25f * red[0] / 524288.f;
}

// ---------------------------------------------------------------------------
// Dilated conv1d (NTC), groupnorm, layernorm, softmax.
// ---------------------------------------------------------------------------
__global__ __launch_bounds__(256)
void conv1d_ntc(const float* __restrict__ x, const float* __restrict__ W,
                const float* __restrict__ bias, float* __restrict__ y, int dil)
{
  const long idx = (long)blockIdx.x * 256 + threadIdx.x;
  const int  o  = (int)(idx & 255);
  const long bt = idx >> 8;
  const int  t  = (int)(bt & 511);
  const int  b  = (int)(bt >> 9);
  float acc = bias[o];
#pragma unroll
  for (int j = 0; j < 3; ++j) {
    const int tt = t + (j - 1) * dil;
    if (tt < 0 || tt >= 512) continue;
    const float* xp = x + ((long)b * 512 + tt) * 256;
    const float* wp = W + (long)o * 768 + j;
    float s = 0.f;
    for (int i = 0; i < 256; ++i) s += xp[i] * wp[i * 3];
    acc += s;
  }
  y[idx] = acc;
}

__global__ __launch_bounds__(1024)
void groupnorm(const float* __restrict__ x, const float* __restrict__ g,
               const float* __restrict__ be, const float* __restrict__ resid,
               float* __restrict__ y, int relu)
{
  __shared__ float s1[1024], s2[1024];
  const int b = blockIdx.x, tid = threadIdx.x;
  const float* xp = x + (long)b * 512 * 256;
  float sum = 0.f, sq = 0.f;
  for (int i = tid; i < 512 * 256; i += 1024) { float v = xp[i]; sum += v; sq += v * v; }
  s1[tid] = sum; s2[tid] = sq; __syncthreads();
  for (int st = 512; st > 0; st >>= 1) {
    if (tid < st) { s1[tid] += s1[tid + st]; s2[tid] += s2[tid + st]; }
    __syncthreads();
  }
  const float mean = s1[0] * (1.f / 131072.f);
  const float inv  = rsqrtf(s2[0] * (1.f / 131072.f) - mean * mean + 1e-5f);
  for (int i = tid; i < 512 * 256; i += 1024) {
    const int c = i & 255;
    float v = (xp[i] - mean) * inv * g[c] + be[c];
    if (resid) v += resid[(long)b * 512 * 256 + i];
    if (relu) v = fmaxf(v, 0.f);
    y[(long)b * 512 * 256 + i] = v;
  }
}

__global__ __launch_bounds__(256)
void layernorm_add(const float* __restrict__ a, const float* __restrict__ b,
                   const float* __restrict__ g, const float* __restrict__ be,
                   float* __restrict__ y)
{
  __shared__ float red[256];
  const long row = blockIdx.x;
  const int  i   = threadIdx.x;
  float v = a[row * 256 + i] + (b ? b[row * 256 + i] : 0.f);
  red[i] = v; __syncthreads();
  for (int st = 128; st > 0; st >>= 1) { if (i < st) red[i] += red[i + st]; __syncthreads(); }
  const float m = red[0] * (1.f / 256.f); __syncthreads();
  const float d = v - m;
  red[i] = d * d; __syncthreads();
  for (int st = 128; st > 0; st >>= 1) { if (i < st) red[i] += red[i + st]; __syncthreads(); }
  const float inv = rsqrtf(red[0] * (1.f / 256.f) + 1e-5f);
  y[row * 256 + i] = d * inv * g[i] + be[i];
}

__global__ __launch_bounds__(512)
void softmax_bias(float* __restrict__ sc, const float* __restrict__ rel)
{
  __shared__ float red[512];
  const long row = blockIdx.x;
  const int  q = (int)(row & 511);
  const int  h = (int)((row >> 9) & 7);
  float* p = sc + row * 512;
  const int k = threadIdx.x;
  int d = k - q; d = d < -512 ? -512 : (d > 512 ? 512 : d);
  const float v = p[k] + rel[(long)(d + 512) * 8 + h];
  red[k] = v; __syncthreads();
  for (int st = 256; st > 0; st >>= 1) { if (k < st) red[k] = fmaxf(red[k], red[k + st]); __syncthreads(); }
  const float mx = red[0]; __syncthreads();
  const float e = __expf(v - mx);
  red[k] = e; __syncthreads();
  for (int st = 256; st > 0; st >>= 1) { if (k < st) red[k] += red[k + st]; __syncthreads(); }
  p[k] = e / red[0];
}

// ---------------------------------------------------------------------------
// Host-side GEMM dispatcher: prefer LDS-staged kernel when shape permits.
// ---------------------------------------------------------------------------
static void launch_gemm(hipStream_t st, int transb, int act,
                        const float* A, int lda, long sAb, long sAh,
                        const float* B, int ldb, long sBb, long sBh,
                        float* C, int ldc, long sCb, long sCh,
                        const float* bias,
                        const float* resid, int ldr, long sRb, long sRh,
                        float scale, int M, int N, int K, int Z, int Hb)
{
  if (transb && (M % 128 == 0) && (N % 64 == 0)) {
    dim3 g(N / 64, M / 128, Z), blk(256);
    if (act == ACT_RELU)
      gemm_wmma_lds<ACT_RELU><<<g, blk, 0, st>>>(A, lda, sAb, sAh, B, ldb, sBb, sBh,
          C, ldc, sCb, sCh, bias, resid, ldr, sRb, sRh, scale, K, Hb);
    else if (act == ACT_GELU)
      gemm_wmma_lds<ACT_GELU><<<g, blk, 0, st>>>(A, lda, sAb, sAh, B, ldb, sBb, sBh,
          C, ldc, sCb, sCh, bias, resid, ldr, sRb, sRh, scale, K, Hb);
    else
      gemm_wmma_lds<ACT_NONE><<<g, blk, 0, st>>>(A, lda, sAb, sAh, B, ldb, sBb, sBh,
          C, ldc, sCb, sCh, bias, resid, ldr, sRb, sRh, scale, K, Hb);
    return;
  }
  dim3 g(N / 32, M / 32, Z), blk(32);
  if (transb) {
    if (act == ACT_RELU)
      gemm_wmma<true, ACT_RELU><<<g, blk, 0, st>>>(A, lda, sAb, sAh, B, ldb, sBb, sBh,
          C, ldc, sCb, sCh, bias, resid, ldr, sRb, sRh, scale, K, Hb);
    else if (act == ACT_GELU)
      gemm_wmma<true, ACT_GELU><<<g, blk, 0, st>>>(A, lda, sAb, sAh, B, ldb, sBb, sBh,
          C, ldc, sCb, sCh, bias, resid, ldr, sRb, sRh, scale, K, Hb);
    else
      gemm_wmma<true, ACT_NONE><<<g, blk, 0, st>>>(A, lda, sAb, sAh, B, ldb, sBb, sBh,
          C, ldc, sCb, sCh, bias, resid, ldr, sRb, sRh, scale, K, Hb);
  } else {
    gemm_wmma<false, ACT_NONE><<<g, blk, 0, st>>>(A, lda, sAb, sAh, B, ldb, sBb, sBh,
        C, ldc, sCb, sCh, bias, resid, ldr, sRb, sRh, scale, K, Hb);
  }
}

extern "C" void kernel_launch(void* const* d_in, const int* in_sizes, int n_in,
                              void* d_out, int out_size, void* d_ws, size_t ws_size,
                              hipStream_t stream)
{
  (void)in_sizes; (void)n_in; (void)out_size; (void)ws_size;
#define IN_F(i) ((const float*)d_in[i])
  // jax tree-flatten (sorted dict keys) order:
  const float* codebook = IN_F(0);
  const float *dW1 = IN_F(1), *dW2 = IN_F(2), *dW3 = IN_F(3);
  const float *db1 = IN_F(4), *db2 = IN_F(5), *db3 = IN_F(6);
  const float *embW = IN_F(7), *embB = IN_F(8);
  const float *eW1 = IN_F(9), *eW2 = IN_F(10), *eb1 = IN_F(11), *eb2 = IN_F(12);
  const float *l_Whh[2][2], *l_Wih[2][2], *l_b[2][2];
  l_Whh[0][1] = IN_F(13); l_Wih[0][1] = IN_F(14); l_b[0][1] = IN_F(15);
  l_Whh[0][0] = IN_F(16); l_Wih[0][0] = IN_F(17); l_b[0][0] = IN_F(18);
  l_Whh[1][1] = IN_F(19); l_Wih[1][1] = IN_F(20); l_b[1][1] = IN_F(21);
  l_Whh[1][0] = IN_F(22); l_Wih[1][0] = IN_F(23); l_b[1][0] = IN_F(24);
  const float *skW = IN_F(25), *skB = IN_F(26);
  const float *x = IN_F(90);

  float* ws    = (float*)d_ws;
  float* h0    = ws;                 // [8192,256]
  float* h1    = ws + 2097152;       // [8192,256]
  float* xs    = ws + 4194304;       // [8192,512] gates / dec r1
  float* qkv   = ws + 8388608;       // [8192,768]
  float* ff    = ws + 14680064;      // [8192,1024]
  float* tmp   = ws + 23068672;      // [8192,256]
  float* tmp2  = ws + 25165824;      // [8192,256]
  float* ze    = ws + 27262976;      // [8192,64]
  float* zq    = ws + 27787264;      // [8192,64]
  int*   idxi  = (int*)(ws + 28311552);
  float* bdist = ws + 28319744;      // [8192]

  float* out_f = (float*)d_out;
  float* recon = out_f;                          // [8192,32]
  float* attn0 = out_f + 262144;                 // [3,16,8,512,512]
  float* lossp = out_f + 262144 + 100663296;     // [1]
  float* idxf  = lossp + 1;                      // [8192]

  const int M = 8192;

  // ---- BiLSTM (2 layers) ----
  const float* cur = x; int din = 32;
  float* louts[2] = { h0, h1 };
  for (int L = 0; L < 2; ++L) {
    for (int d = 0; d < 2; ++d) {
      launch_gemm(stream, 1, ACT_NONE, cur, din, 0, 0, l_Wih[L][d], din, 0, 0,
                  xs, 512, 0, 0, l_b[L][d], nullptr, 0, 0, 0, 1.f, M, 512, din, 1, 1);
      lstm_scan<<<1, 1024, 0, stream>>>(xs, l_Whh[L][d], louts[L] + d * 128, d);
    }
    cur = louts[L]; din = 256;
  }

  // ---- encoder projection ----
  launch_gemm(stream, 1, ACT_RELU, h1, 256, 0, 0, eW1, 256, 0, 0,
              tmp, 256, 0, 0, eb1, nullptr, 0, 0, 0, 1.f, M, 256, 256, 1, 1);
  launch_gemm(stream, 1, ACT_NONE, tmp, 256, 0, 0, eW2, 256, 0, 0,
              ze, 64, 0, 0, eb2, nullptr, 0, 0, 0, 1.f, M, 64, 256, 1, 1);

  // ---- VQ ----
  vq_kernel<<<8192, 256, 0, stream>>>(ze, codebook, zq, idxi, idxf, bdist);
  vq_loss_reduce<<<1, 1024, 0, stream>>>(bdist, lossp);

  // ---- embed ----
  launch_gemm(stream, 1, ACT_NONE, zq, 64, 0, 0, embW, 64, 0, 0,
              h0, 256, 0, 0, embB, nullptr, 0, 0, 0, 1.f, M, 256, 64, 1, 1);

  // ---- TCN (3 blocks, dil = 1,2,4) ----
  for (int i = 0; i < 3; ++i) {
    const int bidx = 27 + i * 8;
    const float *W1 = IN_F(bidx + 0), *W2 = IN_F(bidx + 1);
    const float *b1 = IN_F(bidx + 2), *b2 = IN_F(bidx + 3);
    const float *be1 = IN_F(bidx + 4), *be2 = IN_F(bidx + 5);
    const float *g1 = IN_F(bidx + 6), *g2 = IN_F(bidx + 7);
    const int dil = 1 << i;
    conv1d_ntc<<<8192, 256, 0, stream>>>(h0, W1, b1, tmp, dil);
    groupnorm<<<16, 1024, 0, stream>>>(tmp, g1, be1, nullptr, tmp, 1);
    conv1d_ntc<<<8192, 256, 0, stream>>>(tmp, W2, b2, tmp2, dil);
    groupnorm<<<16, 1024, 0, stream>>>(tmp2, g2, be2, h0, h0, 1);
  }

  // ---- Transformer (3 layers) ----
  for (int l = 0; l < 3; ++l) {
    const int base = 51 + l * 13;
    const float *W1 = IN_F(base + 0), *W2 = IN_F(base + 1), *Wo = IN_F(base + 2),
                *Wqkv = IN_F(base + 3), *b1 = IN_F(base + 4), *b2 = IN_F(base + 5),
                *bo = IN_F(base + 6), *bqkv = IN_F(base + 7),
                *ln1b = IN_F(base + 8), *ln1g = IN_F(base + 9),
                *ln2b = IN_F(base + 10), *ln2g = IN_F(base + 11),
                *rel = IN_F(base + 12);
    float* attn = attn0 + (long)l * 33554432;

    launch_gemm(stream, 1, ACT_NONE, h0, 256, 0, 0, Wqkv, 256, 0, 0,
                qkv, 768, 0, 0, bqkv, nullptr, 0, 0, 0, 1.f, M, 768, 256, 1, 1);
    // scores = Q K^T / sqrt(dh), batched over 128 (b,h) pairs
    launch_gemm(stream, 1, ACT_NONE, qkv, 768, 393216, 32, qkv + 256, 768, 393216, 32,
                attn, 512, 2097152, 262144, nullptr, nullptr, 0, 0, 0,
                0.17677669529663687f, 512, 512, 32, 128, 8);
    softmax_bias<<<65536, 512, 0, stream>>>(attn, rel);
    // ctx = attn @ V  (B is [K,N] row-major -> TRANSB=false)
    launch_gemm(stream, 0, ACT_NONE, attn, 512, 2097152, 262144, qkv + 512, 768, 393216, 32,
                tmp, 256, 131072, 32, nullptr, nullptr, 0, 0, 0,
                1.f, 512, 32, 512, 128, 8);
    // o = ctx @ Wo^T + bo + h
    launch_gemm(stream, 1, ACT_NONE, tmp, 256, 0, 0, Wo, 256, 0, 0,
                tmp2, 256, 0, 0, bo, h0, 256, 0, 0, 1.f, M, 256, 256, 1, 1);
    layernorm_add<<<8192, 256, 0, stream>>>(tmp2, nullptr, ln1g, ln1b, h1);
    launch_gemm(stream, 1, ACT_GELU, h1, 256, 0, 0, W1, 256, 0, 0,
                ff, 1024, 0, 0, b1, nullptr, 0, 0, 0, 1.f, M, 1024, 256, 1, 1);
    launch_gemm(stream, 1, ACT_NONE, ff, 1024, 0, 0, W2, 1024, 0, 0,
                tmp2, 256, 0, 0, b2, h1, 256, 0, 0, 1.f, M, 256, 1024, 1, 1);
    layernorm_add<<<8192, 256, 0, stream>>>(tmp2, nullptr, ln2g, ln2b, h0);
  }

  // ---- decoder + skip ----
  float* r1 = xs;
  launch_gemm(stream, 1, ACT_RELU, h0, 256, 0, 0, dW1, 256, 0, 0,
              r1, 512, 0, 0, db1, nullptr, 0, 0, 0, 1.f, M, 512, 256, 1, 1);
  launch_gemm(stream, 1, ACT_RELU, r1, 512, 0, 0, dW2, 512, 0, 0,
              tmp, 256, 0, 0, db2, nullptr, 0, 0, 0, 1.f, M, 256, 512, 1, 1);
  launch_gemm(stream, 1, ACT_NONE, tmp, 256, 0, 0, dW3, 256, 0, 0,
              tmp2, 32, 0, 0, db3, nullptr, 0, 0, 0, 1.f, M, 32, 256, 1, 1);
  launch_gemm(stream, 1, ACT_NONE, x, 32, 0, 0, skW, 32, 0, 0,
              recon, 32, 0, 0, skB, tmp2, 32, 0, 0, 1.f, M, 32, 32, 1, 1);
#undef IN_F
}
